// NeuralMemory_90615220011251
// MI455X (gfx1250) — compile-verified, matrix operand order checked by reading, not run on hardware
//
#include <hip/hip_runtime.h>
#include <hip/hip_bf16.h>

#define BB 2
#define LL 1024
#define DD 256
#define CC 16
#define NNK 64
#define NCH (BB * NNK)
#define DEPTH 4
#define PP (DEPTH * DD * DD)
#define EPSV 1.1920929e-07f

typedef __attribute__((ext_vector_type(16))) __bf16 v16bf;
typedef __attribute__((ext_vector_type(8)))  float  v8f;
typedef __attribute__((ext_vector_type(4)))  unsigned int v4u;
typedef __attribute__((ext_vector_type(8)))  int v8i_t;
typedef __attribute__((ext_vector_type(4)))  int v4i_t;

// ---------------------------------------------------------------------------
// WMMA helpers (CDNA5 wave32, v_wmma_f32_16x16x32_bf16)
// A fragment: lane l holds row m=l&15; k = (s/8)*16 + (l>>4)*8 + (s&7)
//   -> two contiguous 8-element bf16 runs when storage is k-contiguous.
// B fragment: same index map with "row" = output column (store B transposed).
// C/D: element v of lane l sits at (row = v + 8*(l>>4), col = l&15).
// ---------------------------------------------------------------------------
__device__ __forceinline__ v16bf load_frag(const __bf16* p, int ld, int row0, int k0) {
  const int lane = threadIdx.x & 31;
  const int r = lane & 15, hi = lane >> 4;
  const __bf16* b0 = p + (size_t)(row0 + r) * ld + k0 + hi * 8;
  v16bf f;
#pragma unroll
  for (int j = 0; j < 8; ++j) { f[j] = b0[j]; f[j + 8] = b0[16 + j]; }
  return f;
}

__device__ __forceinline__ v8f wmma_bf16(v16bf a, v16bf b, v8f c) {
  return __builtin_amdgcn_wmma_f32_16x16x32_bf16(false, a, false, b, (short)0, c,
                                                 false, false);
}

__device__ __forceinline__ float sigmoidf_(float x) { return 1.0f / (1.0f + expf(-x)); }

__device__ __forceinline__ float block_sum(float v, float* sRed) {
  const int t = threadIdx.x;
  sRed[t] = v;
  __syncthreads();
#pragma unroll
  for (int s = 128; s > 0; s >>= 1) {
    if (t < s) sRed[t] += sRed[t + s];
    __syncthreads();
  }
  float r = sRed[0];
  __syncthreads();
  return r;
}

// ---------------------------------------------------------------------------
// Tensor Data Mover: 1-row 2D descriptor copying `nbytes` contiguous bytes
// (bf16 elements, data_size=2B) from global memory into LDS. Issue from one
// wave; TDM ignores EXEC. Tracked with TENSORcnt (ISA 08_async_tensor §7-8).
//   group0: count=1 | lds_addr | global_addr[56:0] | type=2
//   group1: data_size=1(2B), tensor_dim0=tile_dim0=nbytes/2, dims1=1
// ---------------------------------------------------------------------------
__device__ __forceinline__ void tdm_load_to_lds_bytes(const void* gsrc, void* lds_dst,
                                                      unsigned int nbytes) {
  const unsigned int elems = nbytes >> 1;                 // bf16 elements
  const unsigned long long ga = (unsigned long long)(size_t)gsrc;
  const unsigned int lds = (unsigned int)(size_t)lds_dst; // addr[31:0] = LDS offset
  v4u g0;
  g0[0] = 1u;                                             // count=1, user mode
  g0[1] = lds;                                            // lds_addr
  g0[2] = (unsigned int)ga;                               // global_addr[31:0]
  g0[3] = (unsigned int)((ga >> 32) & 0x01FFFFFFu) | (2u << 30);  // [56:32]|type=2
  v8i_t g1;
  g1[0] = 0x10000;                                        // data_size = 1 -> 2 bytes
  g1[1] = (int)((elems & 0xFFFFu) << 16);                 // tensor_dim0[15:0]
  g1[2] = (int)(((elems >> 16) & 0xFFFFu) | (1u << 16));  // dim0 hi | tensor_dim1=1
  g1[3] = (int)((elems & 0xFFFFu) << 16);                 // tile_dim0
  g1[4] = 1;                                              // tile_dim1=1, tile_dim2=0
  g1[5] = (int)elems;                                     // tensor_dim0_stride lo
  g1[6] = (int)((elems & 0xFFFFu) << 16);                 // stride0 hi=0 | stride1 lo
  g1[7] = 0;
  v4i_t z = {};
#if __clang_major__ >= 23
  v8i_t z8 = {};
  __builtin_amdgcn_tensor_load_to_lds(g0, g1, z, z, z8, 0);
#else
  __builtin_amdgcn_tensor_load_to_lds(g0, g1, z, z, 0);
#endif
}

// ---------------------------------------------------------------------------
// Kernel 0: build bf16 weight copies.
//   WkvT[n*256+k] = Wkv[k][n]  (n<512)   -- B operand for sn@Wkv
//   WqT [n*256+k] = Wq[k][n]             -- B operand for rn@Wq
//   WT_l[n*256+k] = Wl[k][n]             -- B operand for forward X@W
//   Wrow_l        = Wl (row-major)       -- B operand for dY@W^T
// ---------------------------------------------------------------------------
__global__ void __launch_bounds__(256) convert_kernel(
    const float* __restrict__ Wkv, const float* __restrict__ Wq,
    const float* __restrict__ W0, const float* __restrict__ W1,
    const float* __restrict__ W2, const float* __restrict__ W3,
    __bf16* __restrict__ WkvT, __bf16* __restrict__ WqT,
    __bf16* __restrict__ WTall, __bf16* __restrict__ Wrowall) {
  const int idx = blockIdx.x * 256 + threadIdx.x;
  const float* Ws[4] = {W0, W1, W2, W3};
  if (idx < 2 * DD * DD) {
    int nn = idx >> 8, k = idx & 255;
    WkvT[idx] = (__bf16)Wkv[k * (2 * DD) + nn];
  } else if (idx < 3 * DD * DD) {
    int j = idx - 2 * DD * DD;
    int nn = j >> 8, k = j & 255;
    WqT[j] = (__bf16)Wq[k * DD + nn];
  } else if (idx < 7 * DD * DD) {
    int j = idx - 3 * DD * DD;
    int m = j >> 16, e = j & 65535;
    int nn = e >> 8, k = e & 255;
    WTall[j] = (__bf16)Ws[m][k * DD + nn];
  } else if (idx < 11 * DD * DD) {
    int j = idx - 7 * DD * DD;
    int m = j >> 16, e = j & 65535;
    Wrowall[j] = (__bf16)Ws[m][e];
  }
}

// ---------------------------------------------------------------------------
// Kernel 1: per-chunk prep. One 256-thread block (8 waves) per chunk.
// store-norm -> gates, keys(bf16)/vals(f32) = sn@Wkv, q(bf16) = rn@Wq.
// ---------------------------------------------------------------------------
__global__ void __launch_bounds__(256) prep_kernel(
    const float* __restrict__ seq, const float* __restrict__ wsn,
    const float* __restrict__ wrn, const float* __restrict__ wa,
    const float* __restrict__ wm, const float* __restrict__ wd,
    const __bf16* __restrict__ WkvT, const __bf16* __restrict__ WqT,
    float* __restrict__ vals, float* __restrict__ gates,
    __bf16* __restrict__ keysb, __bf16* __restrict__ qb) {
  __shared__ float  sSeq[CC * DD];
  __shared__ __bf16 sA[CC * DD];
  __shared__ float  sR[CC];
  __shared__ float  sRed[256];

  const int t = threadIdx.x;
  const int ch = blockIdx.x;
  const int b = ch >> 6, n = ch & 63;
  const int lane = t & 31, wave = t >> 5;
  const size_t chb = (size_t)ch * CC * DD;

  // ---- phase 1: store-norm of this chunk ----
  const float* src = seq + ((size_t)b * LL + (size_t)n * CC) * DD;
  for (int i = t; i < CC * DD; i += 256) sSeq[i] = src[i];
  __syncthreads();
  {
    int r = t >> 4, sub = t & 15;
    float ss = 0.f;
#pragma unroll
    for (int i = 0; i < 16; ++i) { float v = sSeq[r * DD + sub + 16 * i]; ss += v * v; }
    sRed[t] = ss;
    __syncthreads();
    if (sub == 0) {
      float tot = 0.f;
#pragma unroll
      for (int i = 0; i < 16; ++i) tot += sRed[(r << 4) + i];
      sR[r] = rsqrtf(tot * (1.0f / DD) + EPSV);
    }
  }
  __syncthreads();
  float cm;
  {
    const int c = t;
    float acc = 0.f;
#pragma unroll
    for (int r = 0; r < CC; ++r) {
      float v = sSeq[r * DD + c] * sR[r] * wsn[c];
      sA[r * DD + c] = (__bf16)v;
      acc += v;
    }
    cm = acc * (1.0f / CC);
  }
  float d0 = block_sum(cm * wa[t], sRed);
  float d1 = block_sum(cm * wm[t], sRed);
  float d2 = block_sum(cm * wd[t], sRed);
  if (t == 0) {
    gates[ch]           = sigmoidf_(d0);
    gates[NCH + ch]     = sigmoidf_(d1);
    gates[2 * NCH + ch] = sigmoidf_(d2);
  }
  __syncthreads();

  // ---- kv = sn @ Wkv : 32 output col-tiles, 4 per wave ----
  for (int tt = 0; tt < 4; ++tt) {
    const int colt = wave * 4 + tt;
    v8f acc = {};
    for (int kk = 0; kk < DD; kk += 32) {
      v16bf a  = load_frag(sA, DD, 0, kk);
      v16bf bf = load_frag(WkvT, DD, colt * 16, kk);
      acc = wmma_bf16(a, bf, acc);
    }
    const int col = colt * 16 + (lane & 15);
    const int hi = lane >> 4;
    if (col < DD) {
#pragma unroll
      for (int v = 0; v < 8; ++v)
        keysb[chb + (size_t)(v + 8 * hi) * DD + col] = (__bf16)acc[v];
    } else {
      const int cv = col - DD;
#pragma unroll
      for (int v = 0; v < 8; ++v)
        vals[chb + (size_t)(v + 8 * hi) * DD + cv] = acc[v];
    }
  }
  __syncthreads();

  // ---- phase 2: retrieve-norm of shifted rows, q = rn @ Wq ----
  for (int i = t; i < CC * DD; i += 256) {
    int r = i >> 8, c = i & 255;
    int g = n * CC + CC - 1 + r;
    sSeq[i] = (g < LL) ? seq[((size_t)b * LL + g) * DD + c] : 0.0f;
  }
  __syncthreads();
  {
    int r = t >> 4, sub = t & 15;
    float ss = 0.f;
#pragma unroll
    for (int i = 0; i < 16; ++i) { float v = sSeq[r * DD + sub + 16 * i]; ss += v * v; }
    sRed[t] = ss;
    __syncthreads();
    if (sub == 0) {
      float tot = 0.f;
#pragma unroll
      for (int i = 0; i < 16; ++i) tot += sRed[(r << 4) + i];
      sR[r] = rsqrtf(tot * (1.0f / DD) + EPSV);
    }
  }
  __syncthreads();
  {
    const int c = t;
#pragma unroll
    for (int r = 0; r < CC; ++r)
      sA[r * DD + c] = (__bf16)(sSeq[r * DD + c] * sR[r] * wrn[c]);
  }
  __syncthreads();
  for (int tt = 0; tt < 2; ++tt) {
    const int colt = wave * 2 + tt;
    v8f acc = {};
    for (int kk = 0; kk < DD; kk += 32) {
      v16bf a  = load_frag(sA, DD, 0, kk);
      v16bf bf = load_frag(WqT, DD, colt * 16, kk);
      acc = wmma_bf16(a, bf, acc);
    }
    const int col = colt * 16 + (lane & 15);
    const int hi = lane >> 4;
#pragma unroll
    for (int v = 0; v < 8; ++v)
      qb[chb + (size_t)(v + 8 * hi) * DD + col] = (__bf16)acc[v];
  }
}

// ---------------------------------------------------------------------------
// Kernel 2: per-chunk MLP gradient -> surprise = -lr * g (written to scanbuf).
// Keys tile staged into LDS by the Tensor Data Mover (overlapped with LDS
// zero-init). LDS: acts (row-major + K=32-padded transposed), silu', dY. 144KB.
// ---------------------------------------------------------------------------
__global__ void __launch_bounds__(256) grad_kernel(
    const __bf16* __restrict__ keysb, const float* __restrict__ vals,
    const float* __restrict__ gates, const __bf16* __restrict__ WTall,
    const __bf16* __restrict__ Wrowall, float* __restrict__ scanbuf) {
  __shared__ __bf16 sX[DEPTH][CC * DD];     // k, silu(h0), silu(h1), silu(h2)
  __shared__ __bf16 sXT[DEPTH][DD * 32];    // transposed, K padded to 32
  __shared__ __bf16 sDer[3][CC * DD];       // silu'(h0..h2)
  __shared__ __bf16 sDY[CC * DD];
  __shared__ __bf16 sDYT[DD * 32];

  const int t = threadIdx.x;
  const int ch = blockIdx.x;
  const int lane = t & 31, wave = t >> 5;
  const size_t chb = (size_t)ch * CC * DD;
  const float lr = gates[ch];

  // TDM: DMA the 8KB keys tile into sX[0] (wave 0 issues; EXEC-independent)
  if (t < 32) tdm_load_to_lds_bytes(keysb + chb, &sX[0][0], CC * DD * 2);
  for (int i = t; i < DEPTH * DD * 32; i += 256) (&sXT[0][0])[i] = (__bf16)0.0f;
  for (int i = t; i < DD * 32; i += 256) sDYT[i] = (__bf16)0.0f;
  if (t < 32) __builtin_amdgcn_s_wait_tensorcnt((short)0);
  __syncthreads();
  for (int i = t; i < CC * DD; i += 256) {
    int r = i >> 8, c = i & 255;
    sXT[0][c * 32 + r] = sX[0][i];
  }
  __syncthreads();

  // -------- forward --------
  for (int l = 0; l < DEPTH; ++l) {
    const __bf16* WTl = WTall + (size_t)l * DD * DD;
    for (int tt = 0; tt < 2; ++tt) {
      const int colt = wave * 2 + tt;
      v8f acc = {};
      for (int kk = 0; kk < DD; kk += 32) {
        v16bf a  = load_frag(sX[l], DD, 0, kk);
        v16bf bf = load_frag(WTl, DD, colt * 16, kk);
        acc = wmma_bf16(a, bf, acc);
      }
      const int col = colt * 16 + (lane & 15);
      const int hi = lane >> 4;
      if (l < 3) {
#pragma unroll
        for (int v = 0; v < 8; ++v) {
          int r = v + 8 * hi;
          float pre = acc[v];
          float sg = sigmoidf_(pre);
          float s = pre * sg;
          float der = sg * (1.0f + pre * (1.0f - sg));
          sX[l + 1][r * DD + col] = (__bf16)s;
          sXT[l + 1][col * 32 + r] = (__bf16)s;
          sDer[l][r * DD + col] = (__bf16)der;
        }
      } else {
#pragma unroll
        for (int v = 0; v < 8; ++v) {
          int r = v + 8 * hi;
          float d = (2.0f / DD) * (acc[v] - vals[chb + (size_t)r * DD + col]);
          sDY[r * DD + col] = (__bf16)d;
          sDYT[col * 32 + r] = (__bf16)d;
        }
      }
    }
    __syncthreads();
  }

  // -------- backward --------
  float* surp = scanbuf + (size_t)ch * PP;
  for (int l = 3; l >= 0; --l) {
    v8f dx0 = {}, dx1 = {};
    if (l > 0) {  // dX = dY @ W_l^T (into registers; sDY stays valid for gW)
      const __bf16* Wr = Wrowall + (size_t)l * DD * DD;
      for (int tt = 0; tt < 2; ++tt) {
        const int colt = wave * 2 + tt;
        v8f acc = {};
        for (int kk = 0; kk < DD; kk += 32) {
          v16bf a  = load_frag(sDY, DD, 0, kk);
          v16bf bf = load_frag(Wr, DD, colt * 16, kk);
          acc = wmma_bf16(a, bf, acc);
        }
        if (tt == 0) dx0 = acc; else dx1 = acc;
      }
    }
    // gW_l = X_l^T @ dY  -> 256 tiles, 32 per wave, one K=32-padded WMMA each
    float* dst = surp + (size_t)l * DD * DD;
    for (int q2 = 0; q2 < 32; ++q2) {
      const int tIdx = wave * 32 + q2;
      const int mt = tIdx >> 4, nt = tIdx & 15;
      v16bf a  = load_frag(sXT[l], 32, mt * 16, 0);
      v16bf bf = load_frag(sDYT, 32, nt * 16, 0);
      v8f acc = {};
      acc = wmma_bf16(a, bf, acc);
      const int col = nt * 16 + (lane & 15);
      const int hi = lane >> 4;
#pragma unroll
      for (int v = 0; v < 8; ++v)
        dst[(size_t)(mt * 16 + v + 8 * hi) * DD + col] = -lr * acc[v];
    }
    __syncthreads();
    if (l > 0) {  // dY <- dX * silu'(h_{l-1})
      for (int tt = 0; tt < 2; ++tt) {
        v8f acc = tt ? dx1 : dx0;
        const int col = (wave * 2 + tt) * 16 + (lane & 15);
        const int hi = lane >> 4;
#pragma unroll
        for (int v = 0; v < 8; ++v) {
          int r = v + 8 * hi;
          float d = acc[v] * (float)sDer[l - 1][r * DD + col];
          sDY[r * DD + col] = (__bf16)d;
          sDYT[col * 32 + r] = (__bf16)d;
        }
      }
      __syncthreads();
    }
  }
}

// ---------------------------------------------------------------------------
// Kernel 3: fused double associative scan, in place (surprise -> updates).
// One thread per (b, p); momentum lives in a register. Prefetch next chunk.
// ---------------------------------------------------------------------------
__global__ void __launch_bounds__(256) scan_kernel(float* __restrict__ buf,
                                                   const float* __restrict__ gates) {
  const size_t idx = (size_t)blockIdx.x * 256 + threadIdx.x;  // [0, B*P)
  const int b = (int)(idx >> 18);                             // P = 2^18
  const size_t p = idx & (PP - 1);
  float m = 0.f, u = 0.f;
  for (int n = 0; n < NNK; ++n) {
    const int ch = b * NNK + n;
    const float am = gates[NCH + ch];
    const float dc = gates[2 * NCH + ch];
    const size_t q = (size_t)ch * PP + p;
    if (n + 1 < NNK) __builtin_prefetch(buf + q + PP, 0, 0);
    const float s = buf[q];
    m = am * m + s;                // momentum scan
    u = (1.0f - dc) * u + m;      // decay scan
    buf[q] = u;
  }
}

// ---------------------------------------------------------------------------
// Kernel 4: retrieve. Per-chunk fast-weight MLP + post-RMSNorm + shift.
// q tile DMA'd into LDS via TDM; sWT (128 KB LDS) holds bf16(W + upd)^T per
// layer and is reused as the f32 output buffer for the post-norm.
// ---------------------------------------------------------------------------
__global__ void __launch_bounds__(256) retrieve_kernel(
    const __bf16* __restrict__ qb, const float* __restrict__ upd,
    const float* __restrict__ W0, const float* __restrict__ W1,
    const float* __restrict__ W2, const float* __restrict__ W3,
    const float* __restrict__ wpn, float* __restrict__ out) {
  __shared__ __bf16 sWT[DD * DD];
  __shared__ __bf16 sAct[2][CC * DD];
  __shared__ float  sR[CC];
  __shared__ float  sRed[256];

  const int t = threadIdx.x;
  const int ch = blockIdx.x;
  const int b = ch >> 6, n = ch & 63;
  const int lane = t & 31, wave = t >> 5;
  const size_t chb = (size_t)ch * CC * DD;
  const float* Wl[DEPTH] = {W0, W1, W2, W3};
  const float* U = upd + (size_t)ch * PP;

  // TDM: DMA the 8KB q tile into sAct[0] while building the first fast weight
  if (t < 32) tdm_load_to_lds_bytes(qb + chb, &sAct[0][0], CC * DD * 2);

  int cur = 0;
  v8f fin0 = {}, fin1 = {};
  for (int l = 0; l < DEPTH; ++l) {
    const float* W  = Wl[l];
    const float* Ul = U + (size_t)l * DD * DD;
    for (int i = t; i < DD * DD; i += 256) {   // build bf16(W+upd)^T
      int k = i >> 8, nn = i & 255;
      sWT[nn * DD + k] = (__bf16)(W[i] + Ul[i]);
    }
    if (l == 0 && t < 32) __builtin_amdgcn_s_wait_tensorcnt((short)0);
    __syncthreads();
    v8f a0 = {}, a1 = {};
    for (int tt = 0; tt < 2; ++tt) {
      const int colt = wave * 2 + tt;
      v8f acc = {};
      for (int kk = 0; kk < DD; kk += 32) {
        v16bf a  = load_frag(sAct[cur], DD, 0, kk);
        v16bf bf = load_frag(sWT, DD, colt * 16, kk);
        acc = wmma_bf16(a, bf, acc);
      }
      if (tt == 0) a0 = acc; else a1 = acc;
    }
    __syncthreads();
    if (l < 3) {
      for (int tt = 0; tt < 2; ++tt) {
        v8f acc = tt ? a1 : a0;
        const int col = (wave * 2 + tt) * 16 + (lane & 15);
        const int hi = lane >> 4;
#pragma unroll
        for (int v = 0; v < 8; ++v) {
          float pre = acc[v];
          sAct[cur ^ 1][(v + 8 * hi) * DD + col] = (__bf16)(pre * sigmoidf_(pre));
        }
      }
      cur ^= 1;
      __syncthreads();
    } else {
      fin0 = a0; fin1 = a1;
    }
  }

  // post RMSNorm + shift by C-1
  float* sOut = (float*)sWT;   // reuse LDS
  for (int tt = 0; tt < 2; ++tt) {
    v8f acc = tt ? fin1 : fin0;
    const int col = (wave * 2 + tt) * 16 + (lane & 15);
    const int hi = lane >> 4;
#pragma unroll
    for (int v = 0; v < 8; ++v) sOut[(v + 8 * hi) * DD + col] = acc[v];
  }
  __syncthreads();
  {
    int r = t >> 4, sub = t & 15;
    float ss = 0.f;
#pragma unroll
    for (int i = 0; i < 16; ++i) { float v = sOut[r * DD + sub + 16 * i]; ss += v * v; }
    sRed[t] = ss;
    __syncthreads();
    if (sub == 0) {
      float tot = 0.f;
#pragma unroll
      for (int i = 0; i < 16; ++i) tot += sRed[(r << 4) + i];
      sR[r] = rsqrtf(tot * (1.0f / DD) + EPSV);
    }
  }
  __syncthreads();
  for (int i = t; i < CC * DD; i += 256) {
    int r = i >> 8, c = i & 255;
    int s = n * CC + r + (CC - 1);
    if (s < LL) out[((size_t)b * LL + s) * DD + c] = sOut[i] * sR[r] * wpn[c];
  }
  if (n == 0) {
    for (int i = t; i < (CC - 1) * DD; i += 256)
      out[(size_t)b * LL * DD + i] = 0.0f;
  }
}

// ---------------------------------------------------------------------------
extern "C" void kernel_launch(void* const* d_in, const int* in_sizes, int n_in,
                              void* d_out, int out_size, void* d_ws, size_t ws_size,
                              hipStream_t stream) {
  (void)in_sizes; (void)n_in; (void)out_size; (void)ws_size;
  const float* seq = (const float*)d_in[0];
  const float* wsn = (const float*)d_in[1];
  const float* wrn = (const float*)d_in[2];
  const float* wpn = (const float*)d_in[3];
  const float* Wq  = (const float*)d_in[4];
  const float* Wkv = (const float*)d_in[5];
  const float* wa  = (const float*)d_in[6];
  const float* wm  = (const float*)d_in[7];
  const float* wd  = (const float*)d_in[8];
  const float* W0  = (const float*)d_in[9];
  const float* W1  = (const float*)d_in[10];
  const float* W2  = (const float*)d_in[11];
  const float* W3  = (const float*)d_in[12];

  char* ws = (char*)d_ws;
  size_t off = 0;
  auto carve = [&](size_t bytes) {
    void* p = ws + off;
    off = (off + bytes + 255) & ~(size_t)255;
    return p;
  };
  float*  vals    = (float*)carve((size_t)NCH * CC * DD * sizeof(float));
  float*  gates   = (float*)carve((size_t)3 * NCH * sizeof(float));
  __bf16* keysb   = (__bf16*)carve((size_t)NCH * CC * DD * sizeof(__bf16));
  __bf16* qb      = (__bf16*)carve((size_t)NCH * CC * DD * sizeof(__bf16));
  __bf16* WkvT    = (__bf16*)carve((size_t)2 * DD * DD * sizeof(__bf16));
  __bf16* WqT     = (__bf16*)carve((size_t)DD * DD * sizeof(__bf16));
  __bf16* WTall   = (__bf16*)carve((size_t)DEPTH * DD * DD * sizeof(__bf16));
  __bf16* Wrowall = (__bf16*)carve((size_t)DEPTH * DD * DD * sizeof(__bf16));
  float*  scanbuf = (float*)carve((size_t)NCH * PP * sizeof(float));  // 128 MB

  convert_kernel<<<(11 * DD * DD) / 256, 256, 0, stream>>>(
      Wkv, Wq, W0, W1, W2, W3, WkvT, WqT, WTall, Wrowall);
  prep_kernel<<<NCH, 256, 0, stream>>>(seq, wsn, wrn, wa, wm, wd, WkvT, WqT,
                                       vals, gates, keysb, qb);
  grad_kernel<<<NCH, 256, 0, stream>>>(keysb, vals, gates, WTall, Wrowall, scanbuf);
  scan_kernel<<<(BB * PP) / 256, 256, 0, stream>>>(scanbuf, gates);
  retrieve_kernel<<<NCH, 256, 0, stream>>>(qb, scanbuf, W0, W1, W2, W3, wpn,
                                           (float*)d_out);
}
